// MyModel_87522843559141
// MI455X (gfx1250) — compile-verified, matrix operand order checked by reading, not run on hardware
//
#include <hip/hip_runtime.h>

typedef unsigned short u16;
typedef __attribute__((ext_vector_type(16))) __bf16 v16bf;
typedef __attribute__((ext_vector_type(8)))  float  v8f;

#define Bb 2048
#define Tt 64
#define Hh 1024
#define Gg 4096   // 4*H

// ---------- bf16 helpers (raw u16 storage, RNE convert) ----------
static __device__ __forceinline__ u16 f2bf(float f) {
    union { float f; unsigned u; } v; v.f = f;
    unsigned u = v.u;
    unsigned r = (u + 0x7FFFu + ((u >> 16) & 1u)) >> 16;
    return (u16)r;
}
static __device__ __forceinline__ float bf2f(u16 h) {
    union { unsigned u; float f; } v; v.u = ((unsigned)h) << 16;
    return v.f;
}
static __device__ __forceinline__ float sigf(float x) {
    return 1.0f / (1.0f + __expf(-x));
}
static __device__ __forceinline__ float tanhfast(float x) {
    // tanh(x) = 1 - 2/(exp(2x)+1)
    return 1.0f - 2.0f / (__expf(2.0f * x) + 1.0f);
}

// ---------- build combined transposed bf16 weight: BT[n*K + k] ----------
// k <  Ipad : W[k, n]   (zero-padded for k >= I)
// k >= Ipad : U[k-Ipad, n]
__global__ void build_BT(u16* __restrict__ BT, const float* __restrict__ W,
                         const float* __restrict__ U, int I, int Ipad, int K) {
    long long idx = (long long)blockIdx.x * blockDim.x + threadIdx.x;
    if (idx >= (long long)Gg * K) return;
    int n = (int)(idx / K), k = (int)(idx % K);
    float v;
    if (k < Ipad) v = (k < I) ? W[(size_t)k * Gg + n] : 0.0f;
    else          v = U[(size_t)(k - Ipad) * Gg + n];
    BT[idx] = f2bf(v);
}

// ---------- pack A = [x_t | h_{t-1}]  (bf16, [B, K]) ----------
__global__ void pack_A(u16* __restrict__ A, const void* __restrict__ in, int in_f32,
                       int I, int Ipad, int K, int t, const u16* __restrict__ hseq) {
    long long idx = (long long)blockIdx.x * blockDim.x + threadIdx.x;
    if (idx >= (long long)Bb * K) return;
    int b = (int)(idx / K), k = (int)(idx % K);
    if (k < Ipad) {
        float v = 0.0f;
        if (k < I) {
            size_t off = (size_t)b * Tt * I + (size_t)t * I + k;
            v = in_f32 ? ((const float*)in)[off] : bf2f(((const u16*)in)[off]);
        }
        A[idx] = f2bf(v);
    } else {
        int h = k - Ipad;
        A[idx] = (t > 0) ? hseq[(size_t)b * Tt * Hh + (size_t)(t - 1) * Hh + h] : (u16)0;
    }
}

// ---------- fused LSTM step: WMMA GEMM + gate nonlinearity + state update ----------
// z[b, g*H + h] = A[b,:] . BT[g*H + h, :]   for g in {i,f,g,o}
// Block tile: 128 batch rows x 32 h-cols x 4 gates. 8 waves: 4 (M) x 2 (h).
// Wave tile: 32 rows x 16 h-cols, with two 16x16 accumulator tiles per gate
// => lane holds matching i/f/g/o elements; gates applied in-register.
// Fragment layouts per CDNA5 ISA 7.12.2.
union Frag { uint4 q[2]; v16bf v; };

__global__ __launch_bounds__(256)
void lstm_step_fused(const u16* __restrict__ A, const u16* __restrict__ BT,
                     const float* __restrict__ bias, float* __restrict__ c,
                     u16* __restrict__ hseq, int K, int t) {
    const int lane = threadIdx.x & 31;
    const int wave = threadIdx.x >> 5;
    const int wm = wave & 3;
    const int wh = wave >> 2;
    const int row0  = blockIdx.y * 128 + wm * 32;   // batch rows
    const int hcol0 = blockIdx.x * 32 + wh * 16;    // hidden columns

    const int r  = lane & 15;          // tile row (A) / weight row (BT)
    const int kb = (lane >> 4) * 8;    // lane K sub-offset

    v8f acc[2][4];
#pragma unroll
    for (int i = 0; i < 2; i++)
#pragma unroll
        for (int g = 0; g < 4; g++) acc[i][g] = (v8f){};

    for (int kk = 0; kk < K; kk += 32) {
        Frag a[2], b[4];
#pragma unroll
        for (int i = 0; i < 2; i++) {
            const u16* p = A + (size_t)(row0 + i * 16 + r) * K + kk + kb;
            a[i].q[0] = *(const uint4*)p;        // K = kb+0..7
            a[i].q[1] = *(const uint4*)(p + 16); // K = kb+16..23
        }
#pragma unroll
        for (int g = 0; g < 4; g++) {
            const u16* p = BT + (size_t)(g * Hh + hcol0 + r) * K + kk + kb;
            b[g].q[0] = *(const uint4*)p;
            b[g].q[1] = *(const uint4*)(p + 16);
        }
#pragma unroll
        for (int i = 0; i < 2; i++)
#pragma unroll
            for (int g = 0; g < 4; g++)
                acc[i][g] = __builtin_amdgcn_wmma_f32_16x16x32_bf16(
                    false, a[i].v, false, b[g].v, (short)0, acc[i][g], false, false);
    }

    // ---- fused epilogue: bias + gates + c/h update ----
    const int col = hcol0 + (lane & 15);
    const int mo  = (lane >> 4) * 8;
    const float bi = bias[col];
    const float bf = bias[Hh + col];
    const float bg = bias[2 * Hh + col];
    const float bo = bias[3 * Hh + col];
#pragma unroll
    for (int i = 0; i < 2; i++)
#pragma unroll
        for (int rg = 0; rg < 8; rg++) {
            const int row = row0 + i * 16 + mo + rg;        // batch index
            float ig = sigf(acc[i][0][rg] + bi);
            float fg = sigf(acc[i][1][rg] + bf);
            float gg = tanhfast(acc[i][2][rg] + bg);
            float og = sigf(acc[i][3][rg] + bo);
            const size_t cidx = (size_t)row * Hh + col;
            float cp = (t > 0) ? c[cidx] : 0.0f;
            float cn = fg * cp + ig * gg;
            float hn = og * tanhfast(cn);
            c[cidx] = cn;
            hseq[(size_t)row * Tt * Hh + (size_t)t * Hh + col] = f2bf(hn);
        }
}

// ---------- g3: LSTM with H=1 (1024 -> 1), one wave per batch row ----------
__global__ void g3_step(const u16* __restrict__ h2seq, const float* __restrict__ W3,
                        const float* __restrict__ U3, const float* __restrict__ b3,
                        float* __restrict__ g3h, float* __restrict__ g3c,
                        float* __restrict__ gen, int t) {
    int b    = blockIdx.x * 8 + (threadIdx.x >> 5);
    int lane = threadIdx.x & 31;
    const u16* hr = h2seq + (size_t)b * Tt * Hh + (size_t)t * Hh;
    float a0 = 0.f, a1 = 0.f, a2 = 0.f, a3 = 0.f;
    for (int k = lane; k < Hh; k += 32) {
        float hv = bf2f(hr[k]);
        float4 w = ((const float4*)W3)[k];       // W3 is [1024,4] row-major
        a0 += hv * w.x; a1 += hv * w.y; a2 += hv * w.z; a3 += hv * w.w;
    }
#pragma unroll
    for (int off = 16; off; off >>= 1) {
        a0 += __shfl_down(a0, off, 32);
        a1 += __shfl_down(a1, off, 32);
        a2 += __shfl_down(a2, off, 32);
        a3 += __shfl_down(a3, off, 32);
    }
    if (lane == 0) {
        float hp = (t > 0) ? g3h[b] : 0.0f;
        float cp = (t > 0) ? g3c[b] : 0.0f;
        a0 += b3[0] + hp * U3[0];
        a1 += b3[1] + hp * U3[1];
        a2 += b3[2] + hp * U3[2];
        a3 += b3[3] + hp * U3[3];
        float ig = sigf(a0), fg = sigf(a1), gg = tanhfast(a2), og = sigf(a3);
        float cn = fg * cp + ig * gg;
        float hn = og * tanhfast(cn);
        g3h[b] = hn; g3c[b] = cn;
        gen[(size_t)b * Tt + t] = hn;
    }
}

// ---------- final dense: disc[row] = dot(h_row, do_W) + do_b (wave per row) ----------
__global__ void dense_out(const u16* __restrict__ dseq, const float* __restrict__ doW,
                          const float* __restrict__ dob, float* __restrict__ disc) {
    int row  = blockIdx.x * 8 + (threadIdx.x >> 5);
    int lane = threadIdx.x & 31;
    const u16* hr = dseq + (size_t)row * Hh;
    float s = 0.0f;
    for (int k = lane; k < Hh; k += 32) s += bf2f(hr[k]) * doW[k];
#pragma unroll
    for (int off = 16; off; off >>= 1) s += __shfl_down(s, off, 32);
    if (lane == 0) disc[row] = s + dob[0];
}

// ---------- host-side driver ----------
static void run_big_layer(const void* in_seq, int in_f32, int I, int Ipad,
                          const u16* BTw, const float* bias,
                          u16* out_seq, u16* Apack, float* cbuf,
                          hipStream_t stream) {
    const int K = Ipad + Hh;
    const long long packN = (long long)Bb * K;
    dim3 ggrid(Hh / 32, Bb / 128); // (32, 16)
    for (int t = 0; t < Tt; t++) {
        pack_A<<<(unsigned)((packN + 255) / 256), 256, 0, stream>>>(
            Apack, in_seq, in_f32, I, Ipad, K, t, out_seq);
        lstm_step_fused<<<ggrid, 256, 0, stream>>>(Apack, BTw, bias, cbuf, out_seq, K, t);
    }
}

extern "C" void kernel_launch(void* const* d_in, const int* in_sizes, int n_in,
                              void* d_out, int out_size, void* d_ws, size_t ws_size,
                              hipStream_t stream) {
    const float* x    = (const float*)d_in[0];
    const float* g1W  = (const float*)d_in[1];
    const float* g1U  = (const float*)d_in[2];
    const float* g1b  = (const float*)d_in[3];
    const float* g2W  = (const float*)d_in[4];
    const float* g2U  = (const float*)d_in[5];
    const float* g2b  = (const float*)d_in[6];
    const float* g3W  = (const float*)d_in[7];
    const float* g3U  = (const float*)d_in[8];
    const float* g3b  = (const float*)d_in[9];
    const float* d1W  = (const float*)d_in[10];
    const float* d1U  = (const float*)d_in[11];
    const float* d1b  = (const float*)d_in[12];
    const float* d2W  = (const float*)d_in[13];
    const float* d2U  = (const float*)d_in[14];
    const float* d2b  = (const float*)d_in[15];
    const float* doW  = (const float*)d_in[16];
    const float* dob  = (const float*)d_in[17];

    float* gen  = (float*)d_out;              // [B,T]
    float* disc = (float*)d_out + (size_t)Bb * Tt;

    // workspace carve-up (256B aligned)
    char* ws = (char*)d_ws;
    auto carve = [&](size_t bytes) {
        char* p = ws;
        ws += (bytes + 255) & ~(size_t)255;
        return (void*)p;
    };
    const int Kg1 = 64 + Hh;    // 1088
    const int Kg2 = Hh + Hh;    // 2048
    const int Kd1 = 32 + Hh;    // 1056 (I=1 padded to 32)
    const int Kd2 = Hh + Hh;    // 2048

    u16*   seqA  = (u16*)  carve((size_t)Bb * Tt * Hh * 2);
    u16*   seqB  = (u16*)  carve((size_t)Bb * Tt * Hh * 2);
    u16*   wg1   = (u16*)  carve((size_t)Gg * Kg1 * 2);
    u16*   wg2   = (u16*)  carve((size_t)Gg * Kg2 * 2);
    u16*   wd1   = (u16*)  carve((size_t)Gg * Kd1 * 2);
    u16*   wd2   = (u16*)  carve((size_t)Gg * Kd2 * 2);
    u16*   Apack = (u16*)  carve((size_t)Bb * 2048 * 2);
    float* cbuf  = (float*)carve((size_t)Bb * Hh * 4);
    float* g3h   = (float*)carve((size_t)Bb * 4);
    float* g3c   = (float*)carve((size_t)Bb * 4);

    // bf16 transposed combined weights (recomputed each call; deterministic)
    {
        long long n;
        n = (long long)Gg * Kg1;
        build_BT<<<(unsigned)((n + 255) / 256), 256, 0, stream>>>(wg1, g1W, g1U, 64, 64, Kg1);
        n = (long long)Gg * Kg2;
        build_BT<<<(unsigned)((n + 255) / 256), 256, 0, stream>>>(wg2, g2W, g2U, Hh, Hh, Kg2);
        n = (long long)Gg * Kd1;
        build_BT<<<(unsigned)((n + 255) / 256), 256, 0, stream>>>(wd1, d1W, d1U, 1, 32, Kd1);
        n = (long long)Gg * Kd2;
        build_BT<<<(unsigned)((n + 255) / 256), 256, 0, stream>>>(wd2, d2W, d2U, Hh, Hh, Kd2);
    }

    // generator
    run_big_layer(x,    1, 64,   64,   wg1, g1b, seqA, Apack, cbuf, stream);
    run_big_layer(seqA, 0, Hh,   Hh,   wg2, g2b, seqB, Apack, cbuf, stream);
    for (int t = 0; t < Tt; t++)
        g3_step<<<Bb / 8 / 32, 256, 0, stream>>>(seqB, g3W, g3U, g3b, g3h, g3c, gen, t);

    // discriminator (seqA free after g2 finished; seqB free after g3 finished)
    run_big_layer(gen,  1, 1,    32,   wd1, d1b, seqA, Apack, cbuf, stream);
    run_big_layer(seqA, 0, Hh,   Hh,   wd2, d2b, seqB, Apack, cbuf, stream);
    dense_out<<<(Bb * Tt) / 8, 256, 0, stream>>>(seqB, doW, dob, disc);
}
